// ContextFreeReasoner_18425409700205
// MI455X (gfx1250) — compile-verified, matrix operand order checked by reading, not run on hardware
//
#include <hip/hip_runtime.h>

typedef _Float16 v16h __attribute__((ext_vector_type(16)));
typedef float    v8f  __attribute__((ext_vector_type(8)));

#define KTOT  4224          // 2E + E*E feature length
#define BROWS 32768

// 16B-chunk loader matching CDNA5 16-bit WMMA A/B fragment layout:
// lane<16 : K offsets {kb+0..7, kb+16..23}; lane>=16 : {kb+8..15, kb+24..31}
union FragU { float4 f4[2]; v16h v; };

__device__ __forceinline__ v16h load_frag(const _Float16* rowptr, int kb, int klo) {
  FragU u;
  u.f4[0] = *reinterpret_cast<const float4*>(rowptr + kb + klo);
  u.f4[1] = *reinterpret_cast<const float4*>(rowptr + kb + klo + 16);
  return u.v;
}

__device__ __forceinline__ v8f wmma_f16(v16h a, v16h b, v8f c) {
  return __builtin_amdgcn_wmma_f32_16x16x32_f16(false, a, false, b, (short)0, c,
                                                false, false);
}

// ---------------- kernel 0: weights f32 -> f16 ----------------
__global__ __launch_bounds__(256) void cvt_kernel(
    const float* __restrict__ andW, const float* __restrict__ subW1,
    _Float16* __restrict__ andW16, _Float16* __restrict__ subW116) {
  int i = blockIdx.x * 256 + threadIdx.x;
  if (i < 64 * KTOT)  andW16[i]  = (_Float16)andW[i];
  if (i < 256 * KTOT) subW116[i] = (_Float16)subW1[i];
}

// ---------------- kernel 1: leaves + NOT + ANY (fp32 VALU) ----------------
__global__ __launch_bounds__(256) void prep_kernel(
    const float* __restrict__ emb,   const float* __restrict__ roleW,
    const float* __restrict__ roleb, const float* __restrict__ notW,
    const float* __restrict__ notb,
    const int* __restrict__ ca, const int* __restrict__ cb,
    const int* __restrict__ cc, const int* __restrict__ ridx,
    _Float16* __restrict__ no16, _Float16* __restrict__ ao16,
    _Float16* __restrict__ c316) {
  __shared__ float x1[4][64];
  __shared__ float x2[4][64];
  const int tid = threadIdx.x;
  const int rl  = tid >> 6;      // 0..3 local row
  const int o   = tid & 63;      // output index
  const int row = blockIdx.x * 4 + rl;

  const int ia = ca[row], ib = cb[row], ic = cc[row], r = ridx[row];
  float c1 = tanhf(emb[(size_t)ia * 64 + o]);
  float c2 = tanhf(emb[(size_t)ib * 64 + o]);
  float c3 = tanhf(emb[(size_t)ic * 64 + o]);
  x1[rl][o] = c1;
  x2[rl][o] = c2;
  c316[(size_t)row * 64 + o] = (_Float16)c3;
  __syncthreads();

  float a1 = notb[o];
  float a2 = roleb[r * 64 + o];
  const float* w1 = notW + o * 64;
  const float* w2 = roleW + ((size_t)r * 64 + o) * 64;
  #pragma unroll 4
  for (int i = 0; i < 64; i += 4) {
    float4 xa = *reinterpret_cast<const float4*>(&x1[rl][i]);
    float4 xb = *reinterpret_cast<const float4*>(&x2[rl][i]);
    float4 wa = *reinterpret_cast<const float4*>(&w1[i]);
    float4 wb = *reinterpret_cast<const float4*>(&w2[i]);
    a1 += wa.x * xa.x + wa.y * xa.y + wa.z * xa.z + wa.w * xa.w;
    a2 += wb.x * xb.x + wb.y * xb.y + wb.z * xb.z + wb.w * xb.w;
  }
  no16[(size_t)row * 64 + o] = (_Float16)tanhf(a1);
  ao16[(size_t)row * 64 + o] = (_Float16)tanhf(a2);
}

// ---------------- kernel 2: AND node fused feature-GEMM + tanh ----------------
// C[B,64] = [no | ao | outer(no,ao)] @ and_W^T
// Per wave: 32 rows (2 row-groups) x 64 cols; each B fragment feeds 2 WMMAs.
__global__ __launch_bounds__(128) void and_gemm_kernel(
    const _Float16* __restrict__ no16, const _Float16* __restrict__ ao16,
    const _Float16* __restrict__ andW16, const float* __restrict__ and_b,
    _Float16* __restrict__ ando16) {
  __shared__ float lds_no[8 * 16 * 65];
  const int tid   = threadIdx.x;
  const int wave  = tid >> 5;       // 0..3
  const int lane  = tid & 31;
  const int lcol  = lane & 15;
  const int lhalf = lane >> 4;
  const int klo   = lhalf * 8;
  const int blockRow = blockIdx.x * 128;

  for (int e = tid; e < 128 * 64; e += 128) {
    int mm = e >> 6, i = e & 63;
    lds_no[(mm >> 4) * (16 * 65) + (mm & 15) * 65 + i] =
        (float)no16[(size_t)(blockRow + mm) * 64 + i];
  }
  __syncthreads();

  const int rowbase = blockRow + wave * 32;   // 2 row-groups per wave
  v16h noF[2][2], aoF[2][2];
  #pragma unroll
  for (int g = 0; g < 2; g++) {
    const _Float16* noRow = no16 + (size_t)(rowbase + g * 16 + lcol) * 64;
    const _Float16* aoRow = ao16 + (size_t)(rowbase + g * 16 + lcol) * 64;
    noF[g][0] = load_frag(noRow, 0, klo);
    noF[g][1] = load_frag(noRow, 32, klo);
    aoF[g][0] = load_frag(aoRow, 0, klo);
    aoF[g][1] = load_frag(aoRow, 32, klo);
  }

  v8f zero = {0.f, 0.f, 0.f, 0.f, 0.f, 0.f, 0.f, 0.f};
  v8f acc[2][4];
  #pragma unroll
  for (int g = 0; g < 2; g++)
    #pragma unroll
    for (int nt = 0; nt < 4; nt++) acc[g][nt] = zero;

  const float* myno0 = &lds_no[(2 * wave + 0) * (16 * 65) + lcol * 65];
  const float* myno1 = &lds_no[(2 * wave + 1) * (16 * 65) + lcol * 65];

  auto mma2 = [&](v16h A0, v16h A1, int kg) {
    #pragma unroll
    for (int nt = 0; nt < 4; nt++) {
      const _Float16* bp = andW16 + (size_t)(nt * 16 + lcol) * KTOT;
      v16h Bf = load_frag(bp, kg, klo);
      acc[0][nt] = wmma_f16(A0, Bf, acc[0][nt]);
      acc[1][nt] = wmma_f16(A1, Bf, acc[1][nt]);
    }
  };

  mma2(noF[0][0], noF[1][0], 0);    // features k =   0..31 : no
  mma2(noF[0][1], noF[1][1], 32);   // features k =  32..63 : no
  mma2(aoF[0][0], aoF[1][0], 64);   // features k =  64..95 : ao
  mma2(aoF[0][1], aoF[1][1], 96);   // features k = 96..127 : ao
  for (int i = 0; i < 64; i++) {    // outer(no,ao): rank-1 blocks
    _Float16 n0 = (_Float16)myno0[i];
    _Float16 n1 = (_Float16)myno1[i];
    int kg = 128 + i * 64;
    mma2(aoF[0][0] * n0, aoF[1][0] * n1, kg);
    mma2(aoF[0][1] * n0, aoF[1][1] * n1, kg + 32);
  }

  #pragma unroll
  for (int g = 0; g < 2; g++) {
    #pragma unroll
    for (int nt = 0; nt < 4; nt++) {
      int col = nt * 16 + lcol;
      float bias = and_b[col];
      #pragma unroll
      for (int r = 0; r < 8; r++) {
        int m = g * 16 + r + lhalf * 8;
        ando16[(size_t)(rowbase + m) * 64 + col] =
            (_Float16)tanhf(acc[g][nt][r] + bias);
      }
    }
  }
}

// ---------------- out init (seed with sub_b2) ----------------
__global__ __launch_bounds__(256) void out_init_kernel(
    float* __restrict__ out, const float* __restrict__ sub_b2) {
  int i = blockIdx.x * 256 + threadIdx.x;
  if (i < BROWS) out[i] = sub_b2[0];
}

// ---------------- kernel 3: SUB node fused feature-GEMM + ELU + dot ----------------
// h = elu([ando | c3 | outer(ando,c3)] @ sub_W1^T + b1); out += h @ sub_W2
__global__ __launch_bounds__(128) void sub_gemm_kernel(
    const _Float16* __restrict__ ando16, const _Float16* __restrict__ c316,
    const _Float16* __restrict__ subW16, const float* __restrict__ sub_b1,
    const float* __restrict__ sub_W2, float* __restrict__ out) {
  __shared__ float lds_ad[8 * 16 * 65];
  const int tid   = threadIdx.x;
  const int wave  = tid >> 5;
  const int lane  = tid & 31;
  const int lcol  = lane & 15;
  const int lhalf = lane >> 4;
  const int klo   = lhalf * 8;
  const int blockRow = blockIdx.x * 128;
  const int colBlock = blockIdx.y * 64;

  for (int e = tid; e < 128 * 64; e += 128) {
    int mm = e >> 6, i = e & 63;
    lds_ad[(mm >> 4) * (16 * 65) + (mm & 15) * 65 + i] =
        (float)ando16[(size_t)(blockRow + mm) * 64 + i];
  }
  __syncthreads();

  const int rowbase = blockRow + wave * 32;
  v16h adF[2][2], c3F[2][2];
  #pragma unroll
  for (int g = 0; g < 2; g++) {
    const _Float16* adRow = ando16 + (size_t)(rowbase + g * 16 + lcol) * 64;
    const _Float16* c3Row = c316  + (size_t)(rowbase + g * 16 + lcol) * 64;
    adF[g][0] = load_frag(adRow, 0, klo);
    adF[g][1] = load_frag(adRow, 32, klo);
    c3F[g][0] = load_frag(c3Row, 0, klo);
    c3F[g][1] = load_frag(c3Row, 32, klo);
  }

  v8f zero = {0.f, 0.f, 0.f, 0.f, 0.f, 0.f, 0.f, 0.f};
  v8f acc[2][4];
  #pragma unroll
  for (int g = 0; g < 2; g++)
    #pragma unroll
    for (int nt = 0; nt < 4; nt++) acc[g][nt] = zero;

  const float* myad0 = &lds_ad[(2 * wave + 0) * (16 * 65) + lcol * 65];
  const float* myad1 = &lds_ad[(2 * wave + 1) * (16 * 65) + lcol * 65];

  auto mma2 = [&](v16h A0, v16h A1, int kg) {
    #pragma unroll
    for (int nt = 0; nt < 4; nt++) {
      const _Float16* bp = subW16 + (size_t)(colBlock + nt * 16 + lcol) * KTOT;
      v16h Bf = load_frag(bp, kg, klo);
      acc[0][nt] = wmma_f16(A0, Bf, acc[0][nt]);
      acc[1][nt] = wmma_f16(A1, Bf, acc[1][nt]);
    }
  };

  mma2(adF[0][0], adF[1][0], 0);
  mma2(adF[0][1], adF[1][1], 32);
  mma2(c3F[0][0], c3F[1][0], 64);
  mma2(c3F[0][1], c3F[1][1], 96);
  for (int i = 0; i < 64; i++) {
    _Float16 a0 = (_Float16)myad0[i];
    _Float16 a1 = (_Float16)myad1[i];
    int kg = 128 + i * 64;
    mma2(c3F[0][0] * a0, c3F[1][0] * a1, kg);
    mma2(c3F[0][1] * a0, c3F[1][1] * a1, kg + 32);
  }

  // epilogue: ELU, dot with sub_W2 over this block's 64 cols, reduce, atomicAdd
  #pragma unroll
  for (int g = 0; g < 2; g++) {
    #pragma unroll
    for (int r = 0; r < 8; r++) {
      float pre = 0.f;
      #pragma unroll
      for (int nt = 0; nt < 4; nt++) {
        int col = colBlock + nt * 16 + lcol;
        float v = acc[g][nt][r] + sub_b1[col];
        float h = v > 0.f ? v : expm1f(v);
        pre += h * sub_W2[col];
      }
      // reduce over the 16 lanes of each half-wave (rows r and r+8)
      pre += __shfl_xor(pre, 1, 32);
      pre += __shfl_xor(pre, 2, 32);
      pre += __shfl_xor(pre, 4, 32);
      pre += __shfl_xor(pre, 8, 32);
      if (lcol == 0) {
        atomicAdd(&out[rowbase + g * 16 + r + lhalf * 8], pre);
      }
    }
  }
}

// ---------------- host launcher ----------------
extern "C" void kernel_launch(void* const* d_in, const int* in_sizes, int n_in,
                              void* d_out, int out_size, void* d_ws, size_t ws_size,
                              hipStream_t stream) {
  (void)in_sizes; (void)n_in; (void)out_size; (void)ws_size;
  const float* emb   = (const float*)d_in[0];
  const float* roleW = (const float*)d_in[1];
  const float* roleb = (const float*)d_in[2];
  const float* notW  = (const float*)d_in[3];
  const float* notb  = (const float*)d_in[4];
  const float* andW  = (const float*)d_in[5];
  const float* andb  = (const float*)d_in[6];
  const float* subW1 = (const float*)d_in[7];
  const float* subb1 = (const float*)d_in[8];
  const float* subW2 = (const float*)d_in[9];
  const float* subb2 = (const float*)d_in[10];
  const int*   ca    = (const int*)d_in[11];
  const int*   cb    = (const int*)d_in[12];
  const int*   cc    = (const int*)d_in[13];
  const int*   ridx  = (const int*)d_in[14];

  char* ws = (char*)d_ws;
  _Float16* no16    = (_Float16*)(ws);                    // 4 MiB
  _Float16* ao16    = (_Float16*)(ws + (size_t)4194304);  // 4 MiB
  _Float16* c316    = (_Float16*)(ws + (size_t)8388608);  // 4 MiB
  _Float16* ando16  = (_Float16*)(ws + (size_t)12582912); // 4 MiB
  _Float16* andW16  = (_Float16*)(ws + (size_t)16777216); // 528 KiB
  _Float16* subW116 = (_Float16*)(ws + (size_t)17317888); // 2.06 MiB
  float* out = (float*)d_out;

  cvt_kernel<<<(256 * KTOT + 255) / 256, 256, 0, stream>>>(andW, subW1, andW16, subW116);
  prep_kernel<<<BROWS / 4, 256, 0, stream>>>(emb, roleW, roleb, notW, notb,
                                             ca, cb, cc, ridx, no16, ao16, c316);
  and_gemm_kernel<<<BROWS / 128, 128, 0, stream>>>(no16, ao16, andW16, andb, ando16);
  out_init_kernel<<<(BROWS + 255) / 256, 256, 0, stream>>>(out, subb2);
  sub_gemm_kernel<<<dim3(BROWS / 128, 4), 128, 0, stream>>>(ando16, c316, subW116,
                                                            subb1, subW2, out);
}
